// SAFCB_6519760355518
// MI455X (gfx1250) — compile-verified, matrix operand order checked by reading, not run on hardware
//
#include <hip/hip_runtime.h>
#include <math.h>
#include <stdint.h>

typedef __attribute__((ext_vector_type(16))) __bf16 v16bf;
typedef __attribute__((ext_vector_type(4)))  __bf16 v4bf;
typedef __attribute__((ext_vector_type(8)))  float  v8f;
typedef __attribute__((ext_vector_type(4)))  unsigned int v4u;
typedef __attribute__((ext_vector_type(8)))  int    v8i;
typedef __attribute__((ext_vector_type(4)))  int    v4i;

#define BN_EPS 1e-5f

#if __has_builtin(__builtin_amdgcn_tensor_load_to_lds)
#define HAVE_TDM 1
#else
#define HAVE_TDM 0
#endif

constexpr int    Bn = 16, Cc = 128, Hh = 128, Ww = 128, Ee = 256, Oo = 128;
constexpr int    HW = Hh * Ww;                       // 16384
constexpr size_t NTOT = (size_t)Bn * Cc * HW;        // 33554432
constexpr float  ORTHO = 0.08838834764831845f;       // 1/sqrt(128)

__device__ __forceinline__ __bf16 f2bf(float f) {
  unsigned u = __float_as_uint(f);
  unsigned r = u + 0x7FFFu + ((u >> 16) & 1u);       // round-to-nearest-even
  unsigned short h = (unsigned short)(r >> 16);
  __bf16 b;
  __builtin_memcpy(&b, &h, sizeof(h));
  return b;
}

// Per-lane K index inside a 32-wide K slice for the 16-bit WMMA A/B layout:
// element e (0..15) of the v16bf fragment, lane-half hh (= lane>>4).
__device__ __forceinline__ int kfrag(int e, int hh) {
  int p = e >> 1, rr = e & 1;
  return (p < 4) ? (2 * p + 8 * hh + rr) : (16 + 2 * (p - 4) + 8 * hh + rr);
}

#if HAVE_TDM
// Issue a 2-D TDM load: tile (tile0 x tile1) of a (dim0 x dim1) bf16 tensor
// (row stride stride0 elements) into LDS at lds_off. D# per ISA 8.3/8.4.
// 6-arg builtin: (uint32x4 g0, int32x8 g1, int32x4, int32x4, int32x8, i32)
__device__ __forceinline__ void tdm_load_2d(const void* gptr, unsigned lds_off,
                                            int dim0, int dim1, int tile0,
                                            int tile1, int stride0) {
  unsigned long long ga = (unsigned long long)(uintptr_t)gptr;
  v4u g0;
  g0[0] = 1u;                                        // count=1, user descriptor
  g0[1] = lds_off;                                   // lds_addr (bytes)
  g0[2] = (unsigned)(ga & 0xFFFFFFFFu);              // global_addr[31:0]
  g0[3] = (unsigned)((ga >> 32) & 0x01FFFFFFu)       // global_addr[56:32]
          | (2u << 30);                              // type = 2 ("image")
  v8i g1;
  g1[0] = (1 << 16);                                 // data_size=1 (2 bytes)
  g1[1] = (dim0 & 0xFFFF) << 16;                     // tensor_dim0[15:0]
  g1[2] = ((dim0 >> 16) & 0xFFFF) | ((dim1 & 0xFFFF) << 16);
  g1[3] = ((dim1 >> 16) & 0xFFFF) | ((tile0 & 0xFFFF) << 16);
  g1[4] = (tile1 & 0xFFFF);                          // tile_dim1, tile_dim2=0
  g1[5] = stride0;                                   // tensor_dim0_stride lo32
  g1[6] = 0;
  g1[7] = 0;
  v4i z4 = {0, 0, 0, 0};
  v8i z8 = {0, 0, 0, 0, 0, 0, 0, 0};
  __builtin_amdgcn_tensor_load_to_lds(g0, g1, z4, z4, z8, 0);
}
#endif

// ---------------------------------------------------------------------------
// 128-point Stockham radix-2 FFT in LDS. 64 threads, one butterfly per thread
// per stage. DIR = -1 forward, +1 inverse. Returns buffer with the result.
// ---------------------------------------------------------------------------
template <int DIR>
__device__ __forceinline__ float2* fft128(float2* a, float2* b, int tid) {
  int n_cur = 128, s_ = 1;
#pragma unroll
  for (int st = 0; st < 7; ++st) {
    int m = n_cur >> 1;
    int p = tid / s_;
    int q = tid - p * s_;
    float theta = 6.283185307179586f * (float)p / (float)n_cur;
    float sn, cs;
    __sincosf(theta, &sn, &cs);
    float wr = cs;
    float wi = (DIR < 0) ? -sn : sn;
    float2 x0 = a[q + s_ * p];
    float2 x1 = a[q + s_ * (p + m)];
    float2 sum = make_float2(x0.x + x1.x, x0.y + x1.y);
    float2 dif = make_float2(x0.x - x1.x, x0.y - x1.y);
    b[q + 2 * s_ * p]       = sum;
    b[q + s_ * (2 * p + 1)] = make_float2(dif.x * wr - dif.y * wi,
                                          dif.x * wi + dif.y * wr);
    __syncthreads();
    float2* t = a; a = b; b = t;
    n_cur >>= 1;
    s_ <<= 1;
  }
  return a;
}

// --------------------------- BN fold into weights ---------------------------
__global__ __launch_bounds__(256) void fold_kernel(
    const float* __restrict__ w1, const float* __restrict__ b1,
    const float* __restrict__ g1, const float* __restrict__ be1,
    const float* __restrict__ m1, const float* __restrict__ v1,
    const float* __restrict__ w2, const float* __restrict__ b2,
    const float* __restrict__ g2, const float* __restrict__ be2,
    const float* __restrict__ m2, const float* __restrict__ v2,
    __bf16* __restrict__ w1f, __bf16* __restrict__ w2f,
    float* __restrict__ bias1f, float* __restrict__ bias2f) {
  __shared__ float s1[Ee];
  __shared__ float s2[Oo];
  const int tid = threadIdx.x;
  if (tid < Ee) {
    float s = g1[tid] * rsqrtf(v1[tid] + BN_EPS);
    s1[tid] = s;
    bias1f[tid] = (b1[tid] - m1[tid]) * s + be1[tid];
  }
  if (tid < Oo) {
    float s = g2[tid] * rsqrtf(v2[tid] + BN_EPS);
    s2[tid] = s;
    bias2f[tid] = (b2[tid] - m2[tid]) * s + be2[tid];
  }
  __syncthreads();
  for (int i = tid; i < Ee * Cc; i += 256) w1f[i] = f2bf(w1[i] * s1[i / Cc]);
  for (int i = tid; i < Oo * Ee; i += 256) w2f[i] = f2bf(w2[i] * s2[i / Ee]);
}

// ------------------------- forward FFT along W (rows) -----------------------
__global__ __launch_bounds__(64) void fft_row_fwd(
    const float* __restrict__ x, float* __restrict__ outRe,
    float* __restrict__ outIm) {
  __shared__ float2 bufA[128], bufB[128];
  const int tid = threadIdx.x;
  const size_t base = (size_t)blockIdx.x * 128;
  bufA[tid]      = make_float2(x[base + tid], 0.0f);
  bufA[tid + 64] = make_float2(x[base + tid + 64], 0.0f);
  __syncthreads();
  float2* r = fft128<-1>(bufA, bufB, tid);
#pragma unroll
  for (int j = 0; j < 2; ++j) {
    int i = tid + j * 64;
    outRe[base + i] = r[i].x * ORTHO;
    outIm[base + i] = r[i].y * ORTHO;
  }
}

// ------------ forward FFT along H (cols) + fftshift(all axes) + HPF ---------
__global__ __launch_bounds__(64) void fft_col_fwd(
    const float* __restrict__ inRe, const float* __restrict__ inIm,
    float* __restrict__ hfr, float* __restrict__ hfi) {
  __shared__ float2 bufA[128], bufB[128];
  const int tid = threadIdx.x;
  const int idx = blockIdx.x;
  const int wq = idx & 127;
  const int c  = (idx >> 7) & 127;
  const int b  = idx >> 14;
  const size_t colbase = ((size_t)(b * Cc + c) * Hh) * Ww + wq;
#pragma unroll
  for (int j = 0; j < 2; ++j) {
    int i = tid + j * 64;
    size_t off = colbase + (size_t)i * Ww;
    bufA[i] = make_float2(inRe[off], inIm[off]);
  }
  __syncthreads();
  float2* r = fft128<-1>(bufA, bufB, tid);
  const int bs = (b + 8) & 15;
  const int cs = (c + 64) & 127;
  const int wsh = (wq + 64) & 127;
  const float dx = (float)wsh - 63.5f;
#pragma unroll
  for (int j = 0; j < 2; ++j) {
    int i = tid + j * 64;
    int hs = (i + 64) & 127;
    float dy = (float)hs - 63.5f;
    float keep = (dy * dy + dx * dx >= 64.0f) ? ORTHO : 0.0f;  // R=8, R^2=64
    size_t doff = (((size_t)(bs * Cc + cs) * Hh + hs)) * Ww + wsh;
    hfr[doff] = r[i].x * keep;
    hfi[doff] = r[i].y * keep;
  }
}

// -------- GEMM1: y = relu(w1f . t + bias1f), t fp32 -> y bf16 (B,E,HW) ------
// Workgroup tile: M=128 (E), N=64 (spatial). A panel (128x128 bf16, 32 KB)
// DMA'd once via TDM; B tile staged transposed (Bs[n][k]) per 32-wide K step
// using float4 global loads along n (2 vector loads per thread per step).
__global__ __launch_bounds__(256) void gemm1_kernel(
    const float* __restrict__ tbase, const __bf16* __restrict__ w1f,
    const float* __restrict__ bias1f, __bf16* __restrict__ ybase) {
  __shared__ __bf16 As[128 * Cc];     // [m][k], full K panel
  __shared__ __bf16 Bs[64 * 32];      // [n][k], transposed
  const int tid = threadIdx.x;
  const int part = blockIdx.z >> 4;
  const int b    = blockIdx.z & 15;
  const int m_base = blockIdx.y * 128;
  const int n_base = blockIdx.x * 64;
  const float* __restrict__ t =
      tbase + (size_t)part * NTOT + (size_t)b * Cc * HW;
  __bf16* __restrict__ y =
      ybase + (size_t)part * ((size_t)Bn * Ee * HW) + (size_t)b * Ee * HW;

  const int wave = tid >> 5, lane = tid & 31;
  const int mi = wave & 3, ni = wave >> 2;        // 4 x 2 wave grid
  const int hh = lane >> 4, l16 = lane & 15;

#if HAVE_TDM
  if (tid < 32) {                                 // wave 0 issues the DMA
    tdm_load_2d(w1f + (size_t)m_base * Cc, (unsigned)(uintptr_t)As,
                /*dim0=*/Cc, /*dim1=*/Ee - m_base,
                /*tile0=*/Cc, /*tile1=*/128, /*stride0=*/Cc);
    __builtin_amdgcn_s_wait_tensorcnt(0);
  }
  __syncthreads();
#else
#pragma unroll
  for (int j = 0; j < 64; ++j) {
    int lin = tid + j * 256;
    As[lin] = w1f[(m_base + (lin >> 7)) * Cc + (lin & 127)];
  }
  __syncthreads();
#endif

  v8f acc00 = {}, acc01 = {}, acc10 = {}, acc11 = {};
  for (int kc = 0; kc < Cc; kc += 32) {
#pragma unroll
    for (int j = 0; j < 2; ++j) {                 // B tile 32k x 64n
      int lin = tid + j * 256;                    // quad index 0..511
      int nq = lin & 15;                          // float4 along n (16/row)
      int kr = lin >> 4;                          // k row 0..31
      float4 f = *(const float4*)&t[(size_t)(kc + kr) * HW + n_base + nq * 4];
      int n = nq * 4;
      Bs[(n + 0) * 32 + kr] = f2bf(f.x);
      Bs[(n + 1) * 32 + kr] = f2bf(f.y);
      Bs[(n + 2) * 32 + kr] = f2bf(f.z);
      Bs[(n + 3) * 32 + kr] = f2bf(f.w);
    }
    if (kc + 32 < Cc)
      __builtin_prefetch(&t[(size_t)(kc + 32) * HW + n_base + (tid & 63)], 0, 1);
    __syncthreads();
    v16bf a0, a1, b0, b1;
#pragma unroll
    for (int e = 0; e < 16; ++e) {
      int k = kfrag(e, hh);
      a0[e] = As[(mi * 32 + l16) * Cc + kc + k];
      a1[e] = As[(mi * 32 + 16 + l16) * Cc + kc + k];
      b0[e] = Bs[(ni * 32 + l16) * 32 + k];
      b1[e] = Bs[(ni * 32 + 16 + l16) * 32 + k];
    }
    acc00 = __builtin_amdgcn_wmma_f32_16x16x32_bf16(false, a0, false, b0,
                                                    (short)0, acc00, false, false);
    acc01 = __builtin_amdgcn_wmma_f32_16x16x32_bf16(false, a0, false, b1,
                                                    (short)0, acc01, false, false);
    acc10 = __builtin_amdgcn_wmma_f32_16x16x32_bf16(false, a1, false, b0,
                                                    (short)0, acc10, false, false);
    acc11 = __builtin_amdgcn_wmma_f32_16x16x32_bf16(false, a1, false, b1,
                                                    (short)0, acc11, false, false);
    __syncthreads();
  }
#pragma unroll
  for (int r = 0; r < 8; ++r) {                   // C layout: M = r + 8*hh
    int e0 = m_base + mi * 32 + r + 8 * hh;
    int e1 = e0 + 16;
    int n0 = n_base + ni * 32 + l16;
    int n1 = n0 + 16;
    float v;
    v = acc00[r] + bias1f[e0]; y[(size_t)e0 * HW + n0] = f2bf(fmaxf(v, 0.0f));
    v = acc01[r] + bias1f[e0]; y[(size_t)e0 * HW + n1] = f2bf(fmaxf(v, 0.0f));
    v = acc10[r] + bias1f[e1]; y[(size_t)e1 * HW + n0] = f2bf(fmaxf(v, 0.0f));
    v = acc11[r] + bias1f[e1]; y[(size_t)e1 * HW + n1] = f2bf(fmaxf(v, 0.0f));
  }
}

// --- GEMM2: z = w2f . y + bias2f + residual(t), y bf16 -> z fp32 (B,O,HW) ---
// Workgroup tile: M=64 (O), N=64. A panel 64x256 bf16 (32 KB) via TDM.
// B staged with bf16x4 (b64) global loads along n.
__global__ __launch_bounds__(256) void gemm2_kernel(
    const __bf16* __restrict__ ybase, const __bf16* __restrict__ w2f,
    const float* __restrict__ bias2f, const float* __restrict__ resbase,
    float* __restrict__ zbase) {
  __shared__ __bf16 As[64 * Ee];      // [m][k], full K panel
  __shared__ __bf16 Bs[64 * 32];      // [n][k], transposed
  const int tid = threadIdx.x;
  const int part = blockIdx.z >> 4;
  const int b    = blockIdx.z & 15;
  const int m_base = blockIdx.y * 64;
  const int n_base = blockIdx.x * 64;
  const __bf16* __restrict__ y =
      ybase + (size_t)part * ((size_t)Bn * Ee * HW) + (size_t)b * Ee * HW;
  const float* __restrict__ res =
      resbase + (size_t)part * NTOT + (size_t)b * Cc * HW;
  float* __restrict__ z =
      zbase + (size_t)part * NTOT + (size_t)b * Oo * HW;

  const int wave = tid >> 5, lane = tid & 31;
  const int mi = wave & 1, ni = wave >> 1;        // 2 x 4 wave grid
  const int hh = lane >> 4, l16 = lane & 15;

#if HAVE_TDM
  if (tid < 32) {
    tdm_load_2d(w2f + (size_t)m_base * Ee, (unsigned)(uintptr_t)As,
                /*dim0=*/Ee, /*dim1=*/Oo - m_base,
                /*tile0=*/Ee, /*tile1=*/64, /*stride0=*/Ee);
    __builtin_amdgcn_s_wait_tensorcnt(0);
  }
  __syncthreads();
#else
#pragma unroll
  for (int j = 0; j < 64; ++j) {
    int lin = tid + j * 256;
    As[lin] = w2f[(m_base + (lin >> 8)) * Ee + (lin & 255)];
  }
  __syncthreads();
#endif

  v8f accA = {}, accB = {};
  for (int kc = 0; kc < Ee; kc += 32) {
#pragma unroll
    for (int j = 0; j < 2; ++j) {                 // B tile 32k x 64n (bf16)
      int lin = tid + j * 256;                    // quad index 0..511
      int nq = lin & 15;
      int kr = lin >> 4;
      v4bf f = *(const v4bf*)&y[(size_t)(kc + kr) * HW + n_base + nq * 4];
      int n = nq * 4;
      Bs[(n + 0) * 32 + kr] = f[0];
      Bs[(n + 1) * 32 + kr] = f[1];
      Bs[(n + 2) * 32 + kr] = f[2];
      Bs[(n + 3) * 32 + kr] = f[3];
    }
    if (kc + 32 < Ee)
      __builtin_prefetch(&y[(size_t)(kc + 32) * HW + n_base + (tid & 63)], 0, 1);
    __syncthreads();
    v16bf a0, a1, bv;
#pragma unroll
    for (int e = 0; e < 16; ++e) {
      int k = kfrag(e, hh);
      a0[e] = As[(mi * 32 + l16) * Ee + kc + k];
      a1[e] = As[(mi * 32 + 16 + l16) * Ee + kc + k];
      bv[e] = Bs[(ni * 16 + l16) * 32 + k];
    }
    accA = __builtin_amdgcn_wmma_f32_16x16x32_bf16(false, a0, false, bv,
                                                   (short)0, accA, false, false);
    accB = __builtin_amdgcn_wmma_f32_16x16x32_bf16(false, a1, false, bv,
                                                   (short)0, accB, false, false);
    __syncthreads();
  }
#pragma unroll
  for (int r = 0; r < 8; ++r) {
    int o0 = m_base + mi * 32 + r + 8 * hh;
    int o1 = o0 + 16;
    int n  = n_base + ni * 16 + l16;
    size_t off0 = (size_t)o0 * HW + n;
    size_t off1 = (size_t)o1 * HW + n;
    z[off0] = accA[r] + bias2f[o0] + res[off0];
    z[off1] = accB[r] + bias2f[o1] + res[off1];
  }
}

// ----------------------- inverse FFT along H (cols) -------------------------
__global__ __launch_bounds__(64) void ifft_col(
    const float* __restrict__ inRe, const float* __restrict__ inIm,
    float* __restrict__ outRe, float* __restrict__ outIm) {
  __shared__ float2 bufA[128], bufB[128];
  const int tid = threadIdx.x;
  const int idx = blockIdx.x;
  const int wq = idx & 127;
  const int c  = (idx >> 7) & 127;
  const int b  = idx >> 14;
  const size_t colbase = ((size_t)(b * Cc + c) * Hh) * Ww + wq;
#pragma unroll
  for (int j = 0; j < 2; ++j) {
    int i = tid + j * 64;
    size_t off = colbase + (size_t)i * Ww;
    bufA[i] = make_float2(inRe[off], inIm[off]);
  }
  __syncthreads();
  float2* r = fft128<1>(bufA, bufB, tid);
#pragma unroll
  for (int j = 0; j < 2; ++j) {
    int i = tid + j * 64;
    size_t off = colbase + (size_t)i * Ww;
    outRe[off] = r[i].x * ORTHO;
    outIm[off] = r[i].y * ORTHO;
  }
}

// ---------------- inverse FFT along W (rows) + |.| -> hx --------------------
__global__ __launch_bounds__(64) void ifft_row_abs(
    const float* __restrict__ inRe, const float* __restrict__ inIm,
    float* __restrict__ hx) {
  __shared__ float2 bufA[128], bufB[128];
  const int tid = threadIdx.x;
  const size_t base = (size_t)blockIdx.x * 128;
#pragma unroll
  for (int j = 0; j < 2; ++j) {
    int i = tid + j * 64;
    bufA[i] = make_float2(inRe[base + i], inIm[base + i]);
  }
  __syncthreads();
  float2* r = fft128<1>(bufA, bufB, tid);
#pragma unroll
  for (int j = 0; j < 2; ++j) {
    int i = tid + j * 64;
    float re = r[i].x * ORTHO, im = r[i].y * ORTHO;
    hx[base + i] = sqrtf(re * re + im * im);
  }
}

// ------------- out1 = x1*hx + x1 ; out2 = x2*hx + x2 (hx in d_out) ----------
__global__ __launch_bounds__(256) void combine_kernel(
    const float* __restrict__ x1, const float* __restrict__ x2,
    float* __restrict__ out) {
  size_t i = ((size_t)blockIdx.x * 256 + threadIdx.x) * 4;
  const float4 a = *(const float4*)(x1 + i);
  const float4 c = *(const float4*)(x2 + i);
  const float4 h = *(const float4*)(out + 2 * NTOT + i);
  float4 o1 = make_float4(a.x * (h.x + 1.0f), a.y * (h.y + 1.0f),
                          a.z * (h.z + 1.0f), a.w * (h.w + 1.0f));
  float4 o2 = make_float4(c.x * (h.x + 1.0f), c.y * (h.y + 1.0f),
                          c.z * (h.z + 1.0f), c.w * (h.w + 1.0f));
  *(float4*)(out + i) = o1;
  *(float4*)(out + NTOT + i) = o2;
}

extern "C" void kernel_launch(void* const* d_in, const int* in_sizes, int n_in,
                              void* d_out, int out_size, void* d_ws,
                              size_t ws_size, hipStream_t stream) {
  (void)in_sizes; (void)n_in; (void)out_size; (void)ws_size;
  const float* x1  = (const float*)d_in[0];
  const float* x2  = (const float*)d_in[1];
  const float* w1  = (const float*)d_in[2];
  const float* b1  = (const float*)d_in[3];
  const float* g1  = (const float*)d_in[4];
  const float* be1 = (const float*)d_in[5];
  const float* m1  = (const float*)d_in[6];
  const float* v1  = (const float*)d_in[7];
  const float* w2  = (const float*)d_in[8];
  const float* b2  = (const float*)d_in[9];
  const float* g2  = (const float*)d_in[10];
  const float* be2 = (const float*)d_in[11];
  const float* m2  = (const float*)d_in[12];
  const float* v2  = (const float*)d_in[13];
  float* out = (float*)d_out;

  // Workspace layout:
  //   [0, 256KB): folded weights/biases
  //   buf0: 2*NTOT fp32 (complex scratch / z)
  //   buf1: 2*NTOT fp32 (hf / ifft intermediate)
  //   ybuf: 2*B*E*HW bf16 (hidden activations, real+imag)
  char* ws = (char*)d_ws;
  __bf16* w1f = (__bf16*)ws;
  __bf16* w2f = w1f + Ee * Cc;
  float* bias1f = (float*)(ws + 2 * (Ee * Cc + Oo * Ee));
  float* bias2f = bias1f + Ee;
  float* buf0 = (float*)(ws + (256 << 10));
  float* buf1 = buf0 + 2 * NTOT;
  __bf16* ybuf = (__bf16*)(buf1 + 2 * NTOT);

  fold_kernel<<<1, 256, 0, stream>>>(w1, b1, g1, be1, m1, v1,
                                     w2, b2, g2, be2, m2, v2,
                                     w1f, w2f, bias1f, bias2f);

  fft_row_fwd<<<Bn * Cc * Hh, 64, 0, stream>>>(x2, buf0, buf0 + NTOT);
  fft_col_fwd<<<Bn * Cc * Ww, 64, 0, stream>>>(buf0, buf0 + NTOT,
                                               buf1, buf1 + NTOT);

  gemm1_kernel<<<dim3(HW / 64, Ee / 128, 2 * Bn), 256, 0, stream>>>(
      buf1, w1f, bias1f, ybuf);
  gemm2_kernel<<<dim3(HW / 64, Oo / 64, 2 * Bn), 256, 0, stream>>>(
      ybuf, w2f, bias2f, buf1, buf0);

  ifft_col<<<Bn * Cc * Ww, 64, 0, stream>>>(buf0, buf0 + NTOT,
                                            buf1, buf1 + NTOT);
  ifft_row_abs<<<Bn * Cc * Hh, 64, 0, stream>>>(buf1, buf1 + NTOT,
                                                out + 2 * NTOT);

  combine_kernel<<<(unsigned)(NTOT / 1024), 256, 0, stream>>>(x1, x2, out);
}
